// FF_19224273617550
// MI455X (gfx1250) — matrix operand order repaired and checked by reading, not compiled, not run on hardware
//
#include <hip/hip_runtime.h>

typedef _Float16 half_t;
typedef _Float16 v16h __attribute__((ext_vector_type(16)));
typedef _Float16 v8h  __attribute__((ext_vector_type(8)));
typedef float    v8f  __attribute__((ext_vector_type(8)));

#define NB    32     // batch
#define LL    512    // sequence length
#define CIN   32
#define DD    512
#define TOPK  6
#define NEG_INF -3.402823466e38f

// ---- WMMA fragment loaders (16x16x32 f16, wave32) -------------------------
// A (16xK=32 slice, row-major, ld elements):
//   lanes 0-15:  row M=lane, halves 0..15 -> K=0..15
//   lanes 16-31: row M=lane-16, K=16..31
__device__ inline v16h loadA16(const half_t* base, int ld, int lane) {
  int m  = lane & 15;
  int k0 = (lane & 16) ? 16 : 0;
  const half_t* p = base + (size_t)m * ld + k0;
  v8h lo = *(const v8h*)(p);
  v8h hi = *(const v8h*)(p + 8);
  v16h a;
#pragma unroll
  for (int j = 0; j < 8; ++j) { a[j] = lo[j]; a[8 + j] = hi[j]; }
  return a;
}

// B (K=32 x 16), sourced from an (N x K) row-major matrix (weightsᵀ GEMM):
//   lane<16: N=lane, K=0..15 ; lane>=16: N=lane-16, K=16..31  (contiguous 32B)
__device__ inline v16h loadBt16(const half_t* baseN, int ld, int lane) {
  int n  = lane & 15;
  int k0 = (lane & 16) ? 16 : 0;
  return *(const v16h*)(baseN + (size_t)n * ld + k0);
}

// ---- Kernel 0: convert wq/wk to f16 ---------------------------------------
__global__ void k_cvt_w(const float* __restrict__ wq, const float* __restrict__ wk,
                        half_t* __restrict__ wq16, half_t* __restrict__ wk16) {
  int i = blockIdx.x * blockDim.x + threadIdx.x;
  if (i < DD * DD) { wq16[i] = (half_t)wq[i]; wk16[i] = (half_t)wk[i]; }
}

// ---- Kernel 1: h = x @ conv_w^T + conv_b  (M=B*L, K=32, N=512) ------------
__global__ void k_h_gemm(const float* __restrict__ x, const float* __restrict__ conv_w,
                         const float* __restrict__ conv_b,
                         float* __restrict__ h_f32, half_t* __restrict__ h_f16) {
  int lane = threadIdx.x & 31;
  int wave = threadIdx.x >> 5;
  int row0 = blockIdx.x * 16;
  int n0   = (blockIdx.y * 4 + wave) * 16;

  // A fragment from x (f32 -> f16 on the fly), K = CIN = 32 exactly
  int m  = lane & 15;
  int ka = (lane & 16) ? 16 : 0;
  const float* px = x + (size_t)(row0 + m) * CIN;
  v16h a;
#pragma unroll
  for (int j = 0; j < 8; ++j) { a[j] = (half_t)px[ka + j]; a[8 + j] = (half_t)px[ka + 8 + j]; }

  // B fragment from conv_w (N x K row-major, f32 -> f16)
  int n  = lane & 15;
  int kb = (lane & 16) ? 16 : 0;
  const float* pw = conv_w + (size_t)(n0 + n) * CIN + kb;
  v16h b;
#pragma unroll
  for (int j = 0; j < 16; ++j) b[j] = (half_t)pw[j];

  v8f c = {};
  c = __builtin_amdgcn_wmma_f32_16x16x32_f16(false, a, false, b, (short)0, c, false, false);

  int nglob = n0 + n;
  float bias = conv_b[nglob];
  int mbase = (lane & 16) ? 8 : 0;
#pragma unroll
  for (int r = 0; r < 8; ++r) {
    int row = row0 + mbase + r;
    float v = c[r] + bias;
    h_f32[(size_t)row * DD + nglob] = v;
    h_f16[(size_t)row * DD + nglob] = (half_t)v;
  }
}

// ---- Kernel 2: q/k = h @ W^T + b  (M=B*L, K=512, N=512; z selects q or k) -
// Register-blocked 16x64 per wave: A fragment loaded once per K step,
// reused across 4 WMMAs (N tiles) to cut VMEM issue per WMMA to ~1.25.
__global__ void k_qk_gemm(const half_t* __restrict__ h16,
                          const half_t* __restrict__ wq16, const half_t* __restrict__ wk16,
                          const float* __restrict__ bq, const float* __restrict__ bk,
                          half_t* __restrict__ q16, half_t* __restrict__ k16o) {
  int lane = threadIdx.x & 31;
  int wave = threadIdx.x >> 5;
  int row0 = blockIdx.x * 16;
  int n0   = blockIdx.y * 256 + wave * 64;       // wave owns 4 N-tiles
  const half_t* W   = (blockIdx.z == 0) ? wq16 : wk16;
  const float* bias = (blockIdx.z == 0) ? bq : bk;
  half_t* out       = (blockIdx.z == 0) ? q16 : k16o;

  v8f acc[4] = {{}, {}, {}, {}};
  for (int kk = 0; kk < DD; kk += 32) {
    v16h a = loadA16(h16 + (size_t)row0 * DD + kk, DD, lane);
#pragma unroll
    for (int j = 0; j < 4; ++j) {
      v16h b = loadBt16(W + (size_t)(n0 + j * 16) * DD + kk, DD, lane);
      acc[j] = __builtin_amdgcn_wmma_f32_16x16x32_f16(false, a, false, b, (short)0, acc[j], false, false);
    }
  }
  int n = lane & 15;
  int mbase = (lane & 16) ? 8 : 0;
#pragma unroll
  for (int j = 0; j < 4; ++j) {
    int nglob = n0 + j * 16 + n;
    float bv = bias[nglob];
#pragma unroll
    for (int r = 0; r < 8; ++r)
      out[(size_t)(row0 + mbase + r) * DD + nglob] = (half_t)(acc[j][r] + bv);
  }
}

// ---- Kernel 3: circular correlation via C = Q K^T, binned by diagonal -----
// mean_corr[b, tau] += sum over (t,s) with (t-s) mod L == tau of <q_t, k_s>
// Register-blocked 16x128 per wave (8 accumulators), K-loop outermost so the
// A fragment (16 q-rows) is loaded once per K step instead of 8x.
__global__ void k_corr(const half_t* __restrict__ q16, const half_t* __restrict__ k16,
                       float* __restrict__ mean_corr) {
  __shared__ float bins[LL];
  int lane = threadIdx.x & 31;
  int wave = threadIdx.x >> 5;
  int t0 = blockIdx.x * 16;
  int b  = blockIdx.y;

  for (int i = threadIdx.x; i < LL; i += blockDim.x) bins[i] = 0.0f;
  __syncthreads();

  const half_t* qb = q16 + (size_t)b * LL * DD;
  const half_t* kb = k16 + (size_t)b * LL * DD;
  int s0 = wave * 128;                            // wave owns 8 s-tiles

  v8f acc[8] = {{}, {}, {}, {}, {}, {}, {}, {}};
  for (int kk = 0; kk < DD; kk += 32) {
    v16h a = loadA16(qb + (size_t)t0 * DD + kk, DD, lane);
#pragma unroll
    for (int i = 0; i < 8; ++i) {
      v16h bm = loadBt16(kb + (size_t)(s0 + i * 16) * DD + kk, DD, lane);
      acc[i] = __builtin_amdgcn_wmma_f32_16x16x32_f16(false, a, false, bm, (short)0, acc[i], false, false);
    }
  }

  int n = lane & 15;
  int mbase = (lane & 16) ? 8 : 0;
#pragma unroll
  for (int i = 0; i < 8; ++i) {
#pragma unroll
    for (int r = 0; r < 8; ++r) {
      int t = t0 + mbase + r;
      int s = s0 + i * 16 + n;
      int tau = (t - s) & (LL - 1);
      atomicAdd(&bins[tau], acc[i][r]);
    }
  }
  __syncthreads();
  for (int i = threadIdx.x; i < LL; i += blockDim.x)
    atomicAdd(&mean_corr[(size_t)b * LL + i], bins[i]);
}

// ---- Kernel 4: top-6 + softmax + v-aggregate + wo + residual + linear head -
__global__ void k_final(const float* __restrict__ h_f32, const float* __restrict__ mean_corr,
                        const float* __restrict__ wv, const float* __restrict__ bv,
                        const float* __restrict__ wo, const float* __restrict__ bo,
                        const float* __restrict__ lin_w, const float* __restrict__ lin_b,
                        float* __restrict__ yout) {
  __shared__ float vals[LL];
  __shared__ float red_v[256];
  __shared__ int   red_i[256];
  __shared__ float topw[TOPK];
  __shared__ int   topi[TOPK];
  __shared__ float hw[DD];
  __shared__ float attn[DD];
  int b = blockIdx.x;
  int tid = threadIdx.x;

  for (int i = tid; i < LL; i += 256)
    vals[i] = mean_corr[(size_t)b * LL + i] * (1.0f / (float)DD);  // mean over H*E
  __syncthreads();

  // top-6 with lowest-index tie break (matches lax.top_k)
  for (int it = 0; it < TOPK; ++it) {
    float bestv = NEG_INF; int besti = 0;
    for (int i = tid; i < LL; i += 256) {
      float v = vals[i];
      if (v > bestv) { bestv = v; besti = i; }
    }
    red_v[tid] = bestv; red_i[tid] = besti;
    __syncthreads();
    for (int off = 128; off > 0; off >>= 1) {
      if (tid < off) {
        float v2 = red_v[tid + off]; int i2 = red_i[tid + off];
        if (v2 > red_v[tid] || (v2 == red_v[tid] && i2 < red_i[tid])) { red_v[tid] = v2; red_i[tid] = i2; }
      }
      __syncthreads();
    }
    if (tid == 0) { topw[it] = red_v[0]; topi[it] = red_i[0]; vals[red_i[0]] = NEG_INF; }
    __syncthreads();
  }

  if (tid == 0) {       // softmax over the 6 selected correlations
    float mx = topw[0];
    for (int k = 1; k < TOPK; ++k) mx = fmaxf(mx, topw[k]);
    float s = 0.f;
    for (int k = 0; k < TOPK; ++k) { topw[k] = __expf(topw[k] - mx); s += topw[k]; }
    for (int k = 0; k < TOPK; ++k) topw[k] /= s;
  }
  __syncthreads();

  // hw[m] = sum_k w_k * h[b, (L-1+delay_k)%L, m]   (only last timestep needed)
  for (int m = tid; m < DD; m += 256) {
    float acc = 0.f;
    for (int k = 0; k < TOPK; ++k) {
      int row = (LL - 1 + topi[k]) & (LL - 1);
      acc += topw[k] * h_f32[((size_t)b * LL + row) * DD + m];
    }
    hw[m] = acc;
  }
  __syncthreads();

  // attn[n] = bv[n] + <hw, wv[n,:]>   (v-projection commutes with the agg sum)
  for (int n = tid; n < DD; n += 256) {
    float acc = bv[n];
    const float* wr = wv + (size_t)n * DD;
    for (int m = 0; m < DD; ++m) acc += hw[m] * wr[m];
    attn[n] = acc;
  }
  __syncthreads();

  // out[n] = h_last[n] + bo[n] + <attn, wo[n,:]> ; y = <out, lin_w> + lin_b
  float ypart = 0.f;
  for (int n = tid; n < DD; n += 256) {
    float acc = bo[n];
    const float* wr = wo + (size_t)n * DD;
    for (int m = 0; m < DD; ++m) acc += attn[m] * wr[m];
    float outv = h_f32[((size_t)b * LL + (LL - 1)) * DD + n] + acc;
    ypart += outv * lin_w[n];
  }
  red_v[tid] = ypart;
  __syncthreads();
  for (int off = 128; off > 0; off >>= 1) {
    if (tid < off) red_v[tid] += red_v[tid + off];
    __syncthreads();
  }
  if (tid == 0) yout[b] = red_v[0] + lin_b[0];
}

extern "C" void kernel_launch(void* const* d_in, const int* in_sizes, int n_in,
                              void* d_out, int out_size, void* d_ws, size_t ws_size,
                              hipStream_t stream) {
  (void)in_sizes; (void)n_in; (void)out_size; (void)ws_size;
  const float* x      = (const float*)d_in[0];
  const float* conv_w = (const float*)d_in[1];
  const float* conv_b = (const float*)d_in[2];
  const float* wq     = (const float*)d_in[3];
  const float* bq     = (const float*)d_in[4];
  const float* wk     = (const float*)d_in[5];
  const float* bk     = (const float*)d_in[6];
  const float* wv     = (const float*)d_in[7];
  const float* bv     = (const float*)d_in[8];
  const float* wo     = (const float*)d_in[9];
  const float* bo     = (const float*)d_in[10];
  const float* lin_w  = (const float*)d_in[11];
  const float* lin_b  = (const float*)d_in[12];

  char* ws = (char*)d_ws;
  float*  h_f32 = (float*) (ws);                    // 33,554,432 B
  half_t* h_f16 = (half_t*)(ws + 33554432);         // 16,777,216 B
  half_t* q16   = (half_t*)(ws + 50331648);         // 16,777,216 B
  half_t* k16   = (half_t*)(ws + 67108864);         // 16,777,216 B
  half_t* wq16  = (half_t*)(ws + 83886080);         //    524,288 B
  half_t* wk16  = (half_t*)(ws + 84410368);         //    524,288 B
  float*  mc    = (float*) (ws + 84934656);         //     65,536 B

  hipMemsetAsync(mc, 0, NB * LL * sizeof(float), stream);
  k_cvt_w<<<(DD * DD + 255) / 256, 256, 0, stream>>>(wq, wk, wq16, wk16);
  k_h_gemm<<<dim3(NB * LL / 16, DD / 64), 128, 0, stream>>>(x, conv_w, conv_b, h_f32, h_f16);
  k_qk_gemm<<<dim3(NB * LL / 16, DD / 256, 2), 128, 0, stream>>>(h_f16, wq16, wk16, bq, bk, q16, k16);
  k_corr<<<dim3(LL / 16, NB), 128, 0, stream>>>(q16, k16, mc);
  k_final<<<NB, 256, 0, stream>>>(h_f32, mc, wv, bv, wo, bo, lin_w, lin_b, (float*)d_out);
}
